// DRGAT_73787538145609
// MI455X (gfx1250) — compile-verified, matrix operand order checked by reading, not run on hardware
//
#include <hip/hip_runtime.h>

typedef __attribute__((ext_vector_type(16))) __bf16 v16bf;
typedef __attribute__((ext_vector_type(8)))  __bf16 v8bf;
typedef __attribute__((ext_vector_type(8)))  float  v8f;

#define NEG_SLOPE 0.2f
#define F 128
#define LNUM 3

// ---------------------------------------------------------------------------
// elementwise f32 -> bf16
__global__ void k_f32_to_bf16(const float* __restrict__ src,
                              __bf16* __restrict__ dst, int n) {
  int i = blockIdx.x * blockDim.x + threadIdx.x;
  if (i < n) dst[i] = (__bf16)src[i];
}

// ---------------------------------------------------------------------------
// Pack B[k][n] = src[k*sk + n*sn] (f32) into the CDNA5 WMMA bf16 B-fragment
// layout: lane = (n%16) + 16*halfsel, element e = 8*hi + lo,
// where k%32 = 16*hi + 8*halfsel + lo.
__global__ void k_pack_b(const float* __restrict__ src, int sk, int sn,
                         __bf16* __restrict__ dst, int K, int N) {
  int t = blockIdx.x * blockDim.x + threadIdx.x;
  if (t >= K * N) return;
  int k = t % K, n = t / K;
  int kt = k >> 5, kin = k & 31;
  int nt = n >> 4, nl = n & 15;
  int hi = kin >> 4, rem = kin & 15;
  int halfsel = rem >> 3, lo = rem & 7;
  int lane = nl + (halfsel << 4);
  int e = (hi << 3) + lo;
  int Ntiles = N >> 4;
  size_t di = ((size_t)(kt * Ntiles + nt) * 32 + lane) * 16 + e;
  dst[di] = (__bf16)src[(size_t)k * sk + (size_t)n * sn];
}

// ---------------------------------------------------------------------------
// Wave32 WMMA GEMM: C[M,N] = A[M,K](bf16,row-major,lda) * Bpacked + bias.
// Each wave computes a 16 x (16*NBT) strip. NBT is compile-time and must
// divide Ntiles exactly, so the K-loop body is completely branch-free:
// 2 A-loads (reused) + NBT B-loads + NBT v_wmma_f32_16x16x32_bf16.
// EXEC stays all-1s around every WMMA (early exit is wave-uniform).
template <int NBT>
__global__ void k_gemm_bf16(const __bf16* __restrict__ A, int lda,
                            const __bf16* __restrict__ Bp,
                            float* __restrict__ C, int ldc,
                            const float* __restrict__ bias,
                            __bf16* __restrict__ Cbf,
                            int Mtiles, int Ntiles, int Ktiles,
                            int streamC) {
  int wid  = (blockIdx.x * blockDim.x + threadIdx.x) >> 5;
  wid = __builtin_amdgcn_readfirstlane(wid);     // wave-uniform by construction
  int lane = threadIdx.x & 31;
  int nGroups = Ntiles / NBT;                    // exact by construction
  if (wid >= Mtiles * nGroups) return;
  int mt  = wid / nGroups;
  int nt0 = (wid % nGroups) * NBT;

  int row   = mt * 16 + (lane & 15);
  int khalf = (lane >> 4) << 3;                  // A fragment K-half select
  v8f c[NBT];
#pragma unroll
  for (int j = 0; j < NBT; ++j) c[j] = (v8f){};

#pragma unroll 1
  for (int kt = 0; kt < Ktiles; ++kt) {
    const __bf16* ap = A + (size_t)row * lda + kt * 32 + khalf;
    v8bf a0 = *(const v8bf*)(ap);                // K = k0 .. k0+7
    v8bf a1 = *(const v8bf*)(ap + 16);           // K = k0+16 .. k0+23
    v16bf a;
#pragma unroll
    for (int e = 0; e < 8; ++e) { a[e] = a0[e]; a[e + 8] = a1[e]; }
    const __bf16* bbase =
        Bp + ((size_t)kt * Ntiles + nt0) * 32 * 16 + (size_t)lane * 16;
#pragma unroll
    for (int j = 0; j < NBT; ++j) {
      v16bf b = *(const v16bf*)(bbase + (size_t)j * 32 * 16);
      c[j] = __builtin_amdgcn_wmma_f32_16x16x32_bf16(
                 false, a, false, b, (short)0, c[j], false, false);
    }
  }

  int rbase = mt * 16 + ((lane >> 4) << 3);
#pragma unroll
  for (int j = 0; j < NBT; ++j) {
    int col  = (nt0 + j) * 16 + (lane & 15);
    float bv = bias ? bias[col] : 0.0f;
#pragma unroll
    for (int v = 0; v < 8; ++v) {
      float val = c[j][v] + bv;
      float* cp = C + (size_t)(rbase + v) * ldc + col;
      if (streamC) __builtin_nontemporal_store(val, cp);
      else         *cp = val;
      if (Cbf) Cbf[(size_t)(rbase + v) * ldc + col] = (__bf16)val;
    }
  }
}

// ---------------------------------------------------------------------------
// Per-node attention scores: s_src[i] = h[i]·a_src, s_dst[i] = h[i]·a_dst.
// One wave per node, 4 floats per lane, xor-shuffle reduction (wave32).
__global__ void k_scores(const float* __restrict__ h,
                         const float* __restrict__ asrc,
                         const float* __restrict__ adst,
                         float* __restrict__ ssrc, float* __restrict__ sdst,
                         int n) {
  int wid  = (blockIdx.x * blockDim.x + threadIdx.x) >> 5;
  int lane = threadIdx.x & 31;
  if (wid >= n) return;
  float4 hv = *(const float4*)(h + (size_t)wid * F + lane * 4);
  float4 as = *(const float4*)(asrc + lane * 4);
  float4 ad = *(const float4*)(adst + lane * 4);
  float s0 = hv.x * as.x + hv.y * as.y + hv.z * as.z + hv.w * as.w;
  float s1 = hv.x * ad.x + hv.y * ad.y + hv.z * ad.z + hv.w * ad.w;
  for (int off = 16; off > 0; off >>= 1) {
    s0 += __shfl_xor(s0, off, 32);
    s1 += __shfl_xor(s1, off, 32);
  }
  if (lane == 0) { ssrc[wid] = s0; sdst[wid] = s1; }
}

// order-preserving float<->uint key for atomic max over signed floats
__device__ __forceinline__ unsigned fkey(float x) {
  unsigned u = __float_as_uint(x);
  return (u & 0x80000000u) ? ~u : (u | 0x80000000u);
}
__device__ __forceinline__ float funkey(unsigned k) {
  return (k & 0x80000000u) ? __uint_as_float(k ^ 0x80000000u)
                           : __uint_as_float(~k);
}

__global__ void k_init_mz(unsigned* __restrict__ mkey, float* __restrict__ z,
                          int n) {
  int i = blockIdx.x * blockDim.x + threadIdx.x;
  if (i < n) { mkey[i] = 0u; z[i] = 0.0f; }
}

// leaky-relu edge score + segment max over dst (self-loops appended at idx>=E)
__global__ void k_edge_max(const int* __restrict__ esrc,
                           const int* __restrict__ edst, int E, int n,
                           const float* __restrict__ ssrc,
                           const float* __restrict__ sdst,
                           float* __restrict__ escore,
                           unsigned* __restrict__ mkey) {
  int i = blockIdx.x * blockDim.x + threadIdx.x;
  if (i >= E + n) return;
  int s = (i < E) ? esrc[i] : (i - E);
  int d = (i < E) ? edst[i] : (i - E);
  float e = ssrc[s] + sdst[d];
  e = (e > 0.0f) ? e : NEG_SLOPE * e;
  escore[i] = e;
  atomicMax(&mkey[d], fkey(e));
}

// w = exp(e - m[dst]); z[dst] += w
__global__ void k_edge_exp(const int* __restrict__ edst, int E, int n,
                           const float* __restrict__ escore,
                           const unsigned* __restrict__ mkey,
                           float* __restrict__ wbuf, float* __restrict__ z) {
  int i = blockIdx.x * blockDim.x + threadIdx.x;
  if (i >= E + n) return;
  int d = (i < E) ? edst[i] : (i - E);
  float w = __expf(escore[i] - funkey(mkey[d]));
  wbuf[i] = w;
  __hip_atomic_fetch_add(&z[d], w, __ATOMIC_RELAXED, __HIP_MEMORY_SCOPE_AGENT);
}

// out[i][f] = bias[f]
__global__ void k_init_out(float* __restrict__ out,
                           const float* __restrict__ bias, int total) {
  int i = blockIdx.x * blockDim.x + threadIdx.x;
  if (i < total) out[i] = bias[i & (F - 1)];
}

// out[dst,:] += (w/z[dst]) * h[src,:]; one wave per edge, 4 floats per lane
__global__ void k_edge_agg(const int* __restrict__ esrc,
                           const int* __restrict__ edst, int E, int n,
                           const float* __restrict__ wbuf,
                           const float* __restrict__ z,
                           const float* __restrict__ h,
                           float* __restrict__ out) {
  int wid  = (blockIdx.x * blockDim.x + threadIdx.x) >> 5;
  int lane = threadIdx.x & 31;
  if (wid >= E + n) return;
  int s = (wid < E) ? esrc[wid] : (wid - E);
  int d = (wid < E) ? edst[wid] : (wid - E);
  float alpha = wbuf[wid] / z[d];
  float4 hv = *(const float4*)(h + (size_t)s * F + lane * 4);
  float* op = out + (size_t)d * F + lane * 4;
  __hip_atomic_fetch_add(op + 0, alpha * hv.x, __ATOMIC_RELAXED, __HIP_MEMORY_SCOPE_AGENT);
  __hip_atomic_fetch_add(op + 1, alpha * hv.y, __ATOMIC_RELAXED, __HIP_MEMORY_SCOPE_AGENT);
  __hip_atomic_fetch_add(op + 2, alpha * hv.z, __ATOMIC_RELAXED, __HIP_MEMORY_SCOPE_AGENT);
  __hip_atomic_fetch_add(op + 3, alpha * hv.w, __ATOMIC_RELAXED, __HIP_MEMORY_SCOPE_AGENT);
}

// Hslice[node*384 + f] = bf16(relu(out[node*128+f]))   (stacked layer output)
__global__ void k_relu_bf16(const float* __restrict__ out,
                            __bf16* __restrict__ Hslice, int total) {
  int i = blockIdx.x * blockDim.x + threadIdx.x;
  if (i >= total) return;
  int node = i >> 7, f = i & (F - 1);
  float v = out[i];
  v = (v > 0.0f) ? v : 0.0f;
  Hslice[(size_t)node * (LNUM * F) + f] = (__bf16)v;
}

// ---------------------------------------------------------------------------
static inline int cdiv(long a, long b) { return (int)((a + b - 1) / b); }

struct Scratch {
  __bf16 *Abf, *Hbf, *Wpack;
  float  *h, *out, *ssrc, *sdst, *z, *escore, *wbuf;
  unsigned *mkey;
};

static void run_branch(const float* x, const int* edges, int E, int n,
                       const float* W, const float* a_src, const float* a_dst,
                       const float* b, const float* Wc, const float* bc,
                       float* Cout, __bf16* Cbf, const Scratch& S,
                       hipStream_t stream) {
  const int BLK = 256;
  const int* esrc = edges;
  const int* edst = edges + E;
  const int tot = E + n;
  const int Mtiles = n / 16;
  const int Ntiles = F / 16;                       // 8 -> NBT=4, exact
  const long waves = (long)Mtiles * (Ntiles / 4);

  k_f32_to_bf16<<<cdiv((long)n * F, BLK), BLK, 0, stream>>>(x, S.Abf, n * F);

  for (int l = 0; l < LNUM; ++l) {
    const float* Wl = W + (size_t)l * F * F;
    // pack W[l] (row-major [K=128][N=128]) into B fragments
    k_pack_b<<<cdiv(F * F, BLK), BLK, 0, stream>>>(Wl, F, 1, S.Wpack, F, F);
    // h = A @ W[l]
    const __bf16* A = (l == 0) ? S.Abf : (S.Hbf + (size_t)(l - 1) * F);
    int lda = (l == 0) ? F : (LNUM * F);
    k_gemm_bf16<4><<<cdiv(waves * 32, BLK), BLK, 0, stream>>>(
        A, lda, S.Wpack, S.h, F, nullptr, nullptr, Mtiles, Ntiles, F / 32, 0);
    // attention scores
    k_scores<<<cdiv((long)n * 32, BLK), BLK, 0, stream>>>(
        S.h, a_src + (size_t)l * F, a_dst + (size_t)l * F, S.ssrc, S.sdst, n);
    // segment softmax over dst
    k_init_mz<<<cdiv(n, BLK), BLK, 0, stream>>>(S.mkey, S.z, n);
    k_edge_max<<<cdiv(tot, BLK), BLK, 0, stream>>>(esrc, edst, E, n, S.ssrc,
                                                   S.sdst, S.escore, S.mkey);
    k_edge_exp<<<cdiv(tot, BLK), BLK, 0, stream>>>(edst, E, n, S.escore,
                                                   S.mkey, S.wbuf, S.z);
    // weighted aggregation + bias
    k_init_out<<<cdiv((long)n * F, BLK), BLK, 0, stream>>>(
        S.out, b + (size_t)l * F, n * F);
    k_edge_agg<<<cdiv((long)tot * 32, BLK), BLK, 0, stream>>>(
        esrc, edst, E, n, S.wbuf, S.z, S.h, S.out);
    // relu -> stacked bf16 buffer [n][3*128]
    k_relu_bf16<<<cdiv((long)n * F, BLK), BLK, 0, stream>>>(
        S.out, S.Hbf + (size_t)l * F, n * F);
  }

  // conv: [n,384] @ B where B[k=l*128+f][e] = Wc[e*384 + k]
  k_pack_b<<<cdiv(LNUM * F * F, BLK), BLK, 0, stream>>>(
      Wc, 1, LNUM * F, S.Wpack, LNUM * F, F);
  k_gemm_bf16<4><<<cdiv(waves * 32, BLK), BLK, 0, stream>>>(
      S.Hbf, LNUM * F, S.Wpack, Cout, F, bc, Cbf, Mtiles, Ntiles,
      (LNUM * F) / 32, 0);
}

extern "C" void kernel_launch(void* const* d_in, const int* in_sizes, int n_in,
                              void* d_out, int out_size, void* d_ws,
                              size_t ws_size, hipStream_t stream) {
  const float* x_m    = (const float*)d_in[0];
  const float* x_d    = (const float*)d_in[1];
  const int*   edges_m = (const int*)d_in[2];
  const int*   edges_d = (const int*)d_in[3];
  const float* Wx     = (const float*)d_in[4];
  const float* ax_src = (const float*)d_in[5];
  const float* ax_dst = (const float*)d_in[6];
  const float* bx     = (const float*)d_in[7];
  const float* Wy     = (const float*)d_in[8];
  const float* ay_src = (const float*)d_in[9];
  const float* ay_dst = (const float*)d_in[10];
  const float* by     = (const float*)d_in[11];
  const float* Wcx    = (const float*)d_in[12];
  const float* bcx    = (const float*)d_in[13];
  const float* Wcy    = (const float*)d_in[14];
  const float* bcy    = (const float*)d_in[15];

  const int n   = in_sizes[0] / F;   // 10000 nodes per branch
  const int E_m = in_sizes[2] / 2;   // 320000
  const int E_d = in_sizes[3] / 2;
  const int Emax = (E_m > E_d) ? E_m : E_d;

  // bump allocator over workspace
  char* wsp = (char*)d_ws;
  auto alloc = [&](size_t bytes) -> char* {
    char* p = wsp;
    wsp += (bytes + 255) & ~(size_t)255;
    return p;
  };
  Scratch S;
  S.Abf    = (__bf16*)alloc((size_t)n * F * 2);
  S.Wpack  = (__bf16*)alloc((size_t)LNUM * F * F * 2);
  S.h      = (float*)alloc((size_t)n * F * 4);
  S.out    = (float*)alloc((size_t)n * F * 4);
  S.ssrc   = (float*)alloc((size_t)n * 4);
  S.sdst   = (float*)alloc((size_t)n * 4);
  S.z      = (float*)alloc((size_t)n * 4);
  S.mkey   = (unsigned*)alloc((size_t)n * 4);
  S.escore = (float*)alloc((size_t)(Emax + n) * 4);
  S.wbuf   = (float*)alloc((size_t)(Emax + n) * 4);
  __bf16* Hbf_m = (__bf16*)alloc((size_t)n * LNUM * F * 2);
  __bf16* Hbf_d = (__bf16*)alloc((size_t)n * LNUM * F * 2);
  float*  Xout  = (float*)alloc((size_t)n * F * 4);
  float*  Yout  = (float*)alloc((size_t)n * F * 4);
  __bf16* Xbf   = (__bf16*)alloc((size_t)n * F * 2);
  __bf16* Ypack = (__bf16*)alloc((size_t)F * n * 2);

  // drug branch -> Xout [n,128] (+ bf16 copy as final-GEMM A operand)
  S.Hbf = Hbf_m;
  run_branch(x_m, edges_m, E_m, n, Wx, ax_src, ax_dst, bx, Wcx, bcx, Xout, Xbf,
             S, stream);
  // disease branch -> Yout [n,128]
  S.Hbf = Hbf_d;
  run_branch(x_d, edges_d, E_d, n, Wy, ay_src, ay_dst, by, Wcy, bcy, Yout,
             nullptr, S, stream);

  // final: Z = X @ Y^T ; B[k][j] = Yout[j*128 + k]
  const int BLK = 256;
  k_pack_b<<<cdiv((long)F * n, BLK), BLK, 0, stream>>>(Yout, 1, F, Ypack, F, n);
  int Mtiles = n / 16, Ntiles = n / 16;          // 625 -> NBT=5, exact (125 groups)
  long waves = (long)Mtiles * (Ntiles / 5);
  k_gemm_bf16<5><<<cdiv(waves * 32, BLK), BLK, 0, stream>>>(
      Xbf, F, Ypack, (float*)d_out, n, nullptr, nullptr, Mtiles, Ntiles,
      F / 32, /*streamC=*/1);
}
